// MirrorAggregator_73907797230010
// MI455X (gfx1250) — matrix operand order repaired and checked by reading, compiled, not run on hardware
//
#include <hip/hip_runtime.h>
#include <hip/hip_bf16.h>
#include <math.h>

#define B_ 512
#define N_ 256
#define D_ 128
#define BN (B_*N_)
#define SCALE 0.08838834764831845f  // 1/sqrt(128)

typedef __bf16 bf16;
typedef bf16   v16bf  __attribute__((ext_vector_type(16)));
typedef bf16   bf16x8 __attribute__((ext_vector_type(8)));
typedef float  v8f    __attribute__((ext_vector_type(8)));
typedef float  f32x4  __attribute__((ext_vector_type(4)));
typedef unsigned int u32x4 __attribute__((ext_vector_type(4)));
typedef int    i32x8  __attribute__((ext_vector_type(8)));
typedef int    i32x4  __attribute__((ext_vector_type(4)));

__device__ __forceinline__ v8f wmma_bf16(v16bf a, v16bf b, v8f c) {
  return __builtin_amdgcn_wmma_f32_16x16x32_bf16(false, a, false, b, (short)0, c, false, false);
}

// Build a WMMA 16-bit B-layout chunk (K p..p+7 and p+16..p+23 at this lane's column)
__device__ __forceinline__ v16bf load_b_chunk(const bf16* p) {
  bf16x8 p0 = *(const bf16x8*)p;
  bf16x8 p1 = *(const bf16x8*)(p + 16);
  v16bf b;
  for (int i = 0; i < 8; ++i) { b[i] = p0[i]; b[8 + i] = p1[i]; }
  return b;
}

// ---------------------------------------------------------------------------
// Kernel 0 (prep, one block):
//   M1h[d][d'] = bf16( sum_e Wq1[e,d] * Wk1[e,d'] )
//   Wq2h / Wk2h = bf16 copies of Wq2 / Wk2 (row-major)
// ---------------------------------------------------------------------------
__global__ __launch_bounds__(256) void k_prep(const float* __restrict__ Wq1,
                                              const float* __restrict__ Wk1,
                                              const float* __restrict__ Wq2,
                                              const float* __restrict__ Wk2,
                                              bf16* __restrict__ M1h,
                                              bf16* __restrict__ Wq2h,
                                              bf16* __restrict__ Wk2h) {
  int lane = threadIdx.x & 31, wave = threadIdx.x >> 5;
  int c16 = lane & 15, half = lane >> 4;
  int drow = wave * 16 + c16;                 // A-matrix row (M index = d)
  v16bf a[4];
  for (int c = 0; c < 4; ++c) {
    for (int i = 0; i < 8; ++i) {
      int e0 = 32 * c + 8 * half + i;
      int e1 = e0 + 16;
      a[c][i]     = (bf16)Wq1[e0 * D_ + drow];   // A[d, e] = Wq1[e, d]
      a[c][8 + i] = (bf16)Wq1[e1 * D_ + drow];
    }
  }
  for (int dt = 0; dt < 8; ++dt) {
    int dcol = dt * 16 + c16;                 // output col = d'
    v8f acc = {};
    for (int c = 0; c < 4; ++c) {
      v16bf b;
      for (int k = 0; k < 16; ++k) {
        int e = 32 * c + 8 * half + k + (k >= 8 ? 8 : 0);
        b[k] = (bf16)Wk1[e * D_ + dcol];      // B[e, d'] = Wk1[e, d']
      }
      acc = wmma_bf16(a[c], b, acc);
    }
    for (int r = 0; r < 8; ++r)
      M1h[(wave * 16 + r + 8 * half) * D_ + dcol] = (bf16)acc[r];
  }
  // pack Wq2 / Wk2 to bf16 (8 floats per thread per iteration)
  for (int i = threadIdx.x * 8; i < D_ * D_; i += 256 * 8) {
    f32x4 q0 = *(const f32x4*)(Wq2 + i);
    f32x4 q1 = *(const f32x4*)(Wq2 + i + 4);
    f32x4 k0 = *(const f32x4*)(Wk2 + i);
    f32x4 k1 = *(const f32x4*)(Wk2 + i + 4);
    bf16x8 qh, kh;
    for (int t = 0; t < 4; ++t) {
      qh[t] = (bf16)q0[t]; qh[4 + t] = (bf16)q1[t];
      kh[t] = (bf16)k0[t]; kh[4 + t] = (bf16)k1[t];
    }
    *(bf16x8*)(Wq2h + i) = qh;
    *(bf16x8*)(Wk2h + i) = kh;
  }
}

// ---------------------------------------------------------------------------
// Kernel 1 (gate+projections): per wave = 16 nodes.
//   T = m @ M1^T ; alpha = <s, T>*scale ; sat_out = s + alpha*(m-s)
//   q2 = m @ Wq2^T ; k2 = sat_out @ Wk2^T ; mirT[b][d][n] = bf16(m)
// ---------------------------------------------------------------------------
__global__ __launch_bounds__(64) void k_gate(const float* __restrict__ mir,
                                             const float* __restrict__ sat,
                                             const bf16* __restrict__ M1h,
                                             const bf16* __restrict__ Wq2h,
                                             const bf16* __restrict__ Wk2h,
                                             float* __restrict__ sat_out,
                                             bf16* __restrict__ q2w,
                                             bf16* __restrict__ k2w,
                                             bf16* __restrict__ mtw) {
  __shared__ float s_s[2][16][D_];
  __shared__ float s_m[2][16][D_];
  __shared__ float s_t[2][16][D_];
  __shared__ float s_al[2][16];
  int lane = threadIdx.x & 31, wave = threadIdx.x >> 5;
  int c16 = lane & 15, half = lane >> 4;
  int g0row = blockIdx.x * 32 + wave * 16;
  int grow  = g0row + c16;                    // this lane's node row
  int b = grow >> 8;
  int n = grow & 255;

  v16bf m_a[4];
  for (int c = 0; c < 4; ++c) {               // load m,s rows -> LDS + bf16 A-regs
    int e0 = 32 * c + 8 * half;
    int e1 = e0 + 16;
    f32x4 m0 = *(const f32x4*)(mir + (size_t)grow * D_ + e0);
    f32x4 m1 = *(const f32x4*)(mir + (size_t)grow * D_ + e0 + 4);
    f32x4 m2 = *(const f32x4*)(mir + (size_t)grow * D_ + e1);
    f32x4 m3 = *(const f32x4*)(mir + (size_t)grow * D_ + e1 + 4);
    f32x4 s0 = *(const f32x4*)(sat + (size_t)grow * D_ + e0);
    f32x4 s1 = *(const f32x4*)(sat + (size_t)grow * D_ + e0 + 4);
    f32x4 s2 = *(const f32x4*)(sat + (size_t)grow * D_ + e1);
    f32x4 s3 = *(const f32x4*)(sat + (size_t)grow * D_ + e1 + 4);
    *(f32x4*)&s_m[wave][c16][e0]     = m0;  *(f32x4*)&s_m[wave][c16][e0 + 4] = m1;
    *(f32x4*)&s_m[wave][c16][e1]     = m2;  *(f32x4*)&s_m[wave][c16][e1 + 4] = m3;
    *(f32x4*)&s_s[wave][c16][e0]     = s0;  *(f32x4*)&s_s[wave][c16][e0 + 4] = s1;
    *(f32x4*)&s_s[wave][c16][e1]     = s2;  *(f32x4*)&s_s[wave][c16][e1 + 4] = s3;
    for (int i = 0; i < 4; ++i) {
      m_a[c][i]      = (bf16)m0[i];  m_a[c][4 + i]  = (bf16)m1[i];
      m_a[c][8 + i]  = (bf16)m2[i];  m_a[c][12 + i] = (bf16)m3[i];
    }
  }
  __syncthreads();

  // T = m @ M1^T  and  q2 = m @ Wq2^T   (bf16 B rows contiguous in memory)
  for (int dt = 0; dt < 8; ++dt) {
    int dcol = dt * 16 + c16;
    v8f accT = {}, accQ = {};
    for (int c = 0; c < 4; ++c) {
      int kb = 32 * c + 8 * half;
      v16bf bT = load_b_chunk(M1h  + (size_t)dcol * D_ + kb);
      v16bf bQ = load_b_chunk(Wq2h + (size_t)dcol * D_ + kb);
      accT = wmma_bf16(m_a[c], bT, accT);
      accQ = wmma_bf16(m_a[c], bQ, accQ);
    }
    for (int r = 0; r < 8; ++r) {
      s_t[wave][r + 8 * half][dcol] = accT[r];
      q2w[(size_t)(g0row + r + 8 * half) * D_ + dcol] = (bf16)accQ[r];
    }
  }
  __syncthreads();

  if (lane < 16) {                             // alpha per row
    float dot = 0.f;
    for (int d = 0; d < D_; ++d) dot += s_s[wave][lane][d] * s_t[wave][lane][d];
    s_al[wave][lane] = dot * SCALE;
  }
  __syncthreads();

  float al = s_al[wave][c16];
  v16bf so_a[4];
  for (int c = 0; c < 4; ++c) {                // sat_out + bf16 A-regs + mirT
    for (int g = 0; g < 2; ++g) {
      int e = 32 * c + 8 * half + g * 16;
      f32x4 sv0 = *(f32x4*)&s_s[wave][c16][e];
      f32x4 sv1 = *(f32x4*)&s_s[wave][c16][e + 4];
      f32x4 mv0 = *(f32x4*)&s_m[wave][c16][e];
      f32x4 mv1 = *(f32x4*)&s_m[wave][c16][e + 4];
      f32x4 o0 = sv0 + al * (mv0 - sv0);
      f32x4 o1 = sv1 + al * (mv1 - sv1);
      *(f32x4*)(sat_out + (size_t)grow * D_ + e)     = o0;
      *(f32x4*)(sat_out + (size_t)grow * D_ + e + 4) = o1;
      for (int i = 0; i < 4; ++i) {
        so_a[c][8 * g + i]     = (bf16)o0[i];
        so_a[c][8 * g + 4 + i] = (bf16)o1[i];
        mtw[(size_t)b * (D_ * N_) + (e + i) * N_ + n]     = (bf16)mv0[i];
        mtw[(size_t)b * (D_ * N_) + (e + 4 + i) * N_ + n] = (bf16)mv1[i];
      }
    }
  }

  // k2 = sat_out @ Wk2^T
  for (int dt = 0; dt < 8; ++dt) {
    int dcol = dt * 16 + c16;
    v8f acc = {};
    for (int c = 0; c < 4; ++c) {
      int kb = 32 * c + 8 * half;
      v16bf bK = load_b_chunk(Wk2h + (size_t)dcol * D_ + kb);
      acc = wmma_bf16(so_a[c], bK, acc);
    }
    for (int r = 0; r < 8; ++r)
      k2w[(size_t)(g0row + r + 8 * half) * D_ + dcol] = (bf16)acc[r];
  }
}

// ---------------------------------------------------------------------------
// Kernel 2: masked softmax attention. Block = 4 waves = 64 query rows of one
// batch. k2[b] (64KB) and mirT[b] (64KB) staged into LDS via the TDM.
// ---------------------------------------------------------------------------
__global__ __launch_bounds__(128) void k_attn(const bf16* __restrict__ q2w,
                                              const bf16* __restrict__ k2w,
                                              const bf16* __restrict__ mtw,
                                              const int*  __restrict__ mask,
                                              float* __restrict__ mir_out) {
  extern __shared__ char dyn[];
  bf16*  k2s  = (bf16*)dyn;                  // [256][128] bf16  (64KB @ 0)
  bf16*  mts  = (bf16*)(dyn + 65536);        // [128][256] bf16  (64KB @ 64K)
  float* scs  = (float*)(dyn + 131072);      // [4][16][256] f32 scores
  bf16*  prs  = (bf16*)(dyn + 196608);       // [4][16][256] bf16 probs
  float* rss  = (float*)(dyn + 229376);      // [4][16] reciprocal sums

  int lane = threadIdx.x & 31, wave = threadIdx.x >> 5;
  int c16 = lane & 15, half = lane >> 4;
  int qbase = blockIdx.x * 64 + wave * 16;
  int b = (blockIdx.x * 64) >> 8;

#if __has_builtin(__builtin_amdgcn_tensor_load_to_lds) && __has_builtin(__builtin_amdgcn_s_wait_tensorcnt)
  if (threadIdx.x == 0) {
    // D# group1: data_size=2B, 1-D tensor: dim0=tile0=stride0=32768 elems
    i32x8 g1; i32x4 z4 = {}; i32x8 z8 = {};
    g1[0] = 0x00010000;           // data_size=1 (2 bytes)
    g1[1] = (int)0x80000000u;     // tensor_dim0 = 32768 (bits 79:48 low half)
    g1[2] = 0x00010000;           // tensor_dim1 = 1
    g1[3] = (int)0x80000000u;     // tile_dim0 = 32768
    g1[4] = 0; g1[5] = 32768; g1[6] = 0; g1[7] = 0;  // dim0_stride=32768
    unsigned long long ga = (unsigned long long)(k2w + (size_t)b * N_ * D_);
    u32x4 g0;
    g0[0] = 1u;                   // count=1, user descriptor
    g0[1] = 0u;                   // lds_addr = 0
    g0[2] = (unsigned)ga;
    g0[3] = (unsigned)(ga >> 32) | 0x80000000u;   // type=2 ("image")
    __builtin_amdgcn_tensor_load_to_lds(g0, g1, z4, z4, z8, 0);
    unsigned long long gb = (unsigned long long)(mtw + (size_t)b * D_ * N_);
    g0[1] = 65536u;               // lds_addr = 64KB
    g0[2] = (unsigned)gb;
    g0[3] = (unsigned)(gb >> 32) | 0x80000000u;
    __builtin_amdgcn_tensor_load_to_lds(g0, g1, z4, z4, z8, 0);
  }
  __builtin_amdgcn_s_wait_tensorcnt(0);
#else
  {
    const bf16x8* s1 = (const bf16x8*)(k2w + (size_t)b * N_ * D_);
    const bf16x8* s2 = (const bf16x8*)(mtw + (size_t)b * D_ * N_);
    bf16x8* d1 = (bf16x8*)k2s;
    bf16x8* d2 = (bf16x8*)mts;
    for (int i = threadIdx.x; i < (N_ * D_) / 8; i += blockDim.x) {
      d1[i] = s1[i];
      d2[i] = s2[i];
    }
  }
#endif
  __syncthreads();

  // q2 A-regs for this wave's 16 query rows
  v16bf aq[4];
  {
    const bf16* qp = q2w + (size_t)(qbase + c16) * D_;
    for (int c = 0; c < 4; ++c) {
      int e0 = 32 * c + 8 * half;
      bf16x8 g0v = *(const bf16x8*)(qp + e0);
      bf16x8 g1v = *(const bf16x8*)(qp + e0 + 16);
      for (int i = 0; i < 8; ++i) { aq[c][i] = g0v[i]; aq[c][8 + i] = g1v[i]; }
    }
  }

  const int* mrow = mask + b * N_;
  float* myscs = scs + wave * 16 * N_;
  bf16*  myprs = prs + wave * 16 * N_;
  float* myrss = rss + wave * 16;

  // scores: S[i,j] = q2[i]·k2[j] * scale, masked
  for (int jt = 0; jt < 16; ++jt) {
    int j = jt * 16 + c16;
    v8f acc = {};
    for (int c = 0; c < 4; ++c) {
      int kb = 32 * c + 8 * half;
      v16bf bk = load_b_chunk(k2s + j * D_ + kb);
      acc = wmma_bf16(aq[c], bk, acc);
    }
    int mj = mrow[j];
    for (int r = 0; r < 8; ++r)
      myscs[(r + 8 * half) * N_ + j] = mj ? acc[r] * SCALE : -3.0e38f;
  }
  __syncthreads();

  if (lane < 16) {                             // row softmax (two-pass)
    float mx = -3.0e38f;
    for (int j = 0; j < N_; ++j) mx = fmaxf(mx, myscs[lane * N_ + j]);
    float sum = 0.f;
    for (int j = 0; j < N_; ++j) {
      float p = __expf(myscs[lane * N_ + j] - mx);
      sum += p;
      myprs[lane * N_ + j] = (bf16)p;
    }
    myrss[lane] = 1.0f / sum;
  }
  __syncthreads();

  // out = P @ mirror   (contraction over keys, 32 per chunk)
  v8f accO[8] = {};
  for (int jc = 0; jc < 8; ++jc) {
    int j0 = 32 * jc + 8 * half;
    bf16x8 p0 = *(const bf16x8*)(myprs + c16 * N_ + j0);
    bf16x8 p1 = *(const bf16x8*)(myprs + c16 * N_ + j0 + 16);
    v16bf ap;
    for (int i = 0; i < 8; ++i) { ap[i] = p0[i]; ap[8 + i] = p1[i]; }
    int kb = 32 * jc + 8 * half;
    for (int dt = 0; dt < 8; ++dt) {
      int d = dt * 16 + c16;
      v16bf bm = load_b_chunk(mts + d * N_ + kb);
      accO[dt] = wmma_bf16(ap, bm, accO[dt]);
    }
  }
  for (int dt = 0; dt < 8; ++dt) {
    int d = dt * 16 + c16;
    for (int r = 0; r < 8; ++r) {
      int row = r + 8 * half;
      mir_out[(size_t)(qbase + row) * D_ + d] = accO[dt][r] * myrss[row];
    }
  }
}

// ---------------------------------------------------------------------------
extern "C" void kernel_launch(void* const* d_in, const int* in_sizes, int n_in,
                              void* d_out, int out_size, void* d_ws, size_t ws_size,
                              hipStream_t stream) {
  (void)in_sizes; (void)n_in; (void)out_size; (void)ws_size;
  const float* mir = (const float*)d_in[0];
  const float* sat = (const float*)d_in[1];
  const int*  mask = (const int*)d_in[2];
  const float* Wq1 = (const float*)d_in[3];
  const float* Wk1 = (const float*)d_in[4];
  const float* Wq2 = (const float*)d_in[5];
  const float* Wk2 = (const float*)d_in[6];

  float* sat_out = (float*)d_out;
  float* mir_out = (float*)d_out + (size_t)BN * D_;

  char* ws = (char*)d_ws;
  bf16* M1h  = (bf16*)ws;                                   // 32 KB
  bf16* Wq2h = (bf16*)(ws + 32768);                         // 32 KB
  bf16* Wk2h = (bf16*)(ws + 65536);                         // 32 KB
  bf16* q2w  = (bf16*)(ws + 131072);                        // 32 MB
  bf16* k2w  = (bf16*)(ws + 131072 + (size_t)BN * D_ * 2);  // 32 MB
  bf16* mtw  = (bf16*)(ws + 131072 + (size_t)BN * D_ * 4);  // 32 MB

  k_prep<<<1, 256, 0, stream>>>(Wq1, Wk1, Wq2, Wk2, M1h, Wq2h, Wk2h);
  k_gate<<<BN / 32, 64, 0, stream>>>(mir, sat, M1h, Wq2h, Wk2h,
                                     sat_out, q2w, k2w, mtw);
  size_t dynbytes = 229632;  // k2 64K + mirT 64K + scores 64K + probs 32K + sums
  k_attn<<<BN / 64, 128, dynbytes, stream>>>(q2w, k2w, mtw, mask, mir_out);
}
